// UNet3D_GeoSample_78013785964616
// MI455X (gfx1250) — matrix operand order repaired and checked by reading, not compiled, not run on hardware
//
#include <hip/hip_runtime.h>

// ---------------------------------------------------------------------------
// CDNA5 (gfx1250) GeoSample 3D U-Net forward pass.
// Activations: channel-major f32 [C][Z][Y][X] between stages.
// GEMM inputs: f16, weights padded [M][Kpad], activations channel-last [N][Kpad].
// All 1x1 convs run through v_wmma_f32_16x16x32_f16 (f32 accumulate).
// ---------------------------------------------------------------------------

typedef __attribute__((ext_vector_type(16))) _Float16 v16h;
typedef __attribute__((ext_vector_type(8)))  _Float16 v8h;
typedef __attribute__((ext_vector_type(8)))  float    v8f;

#define LRELU_SLOPE 0.01f
#define IN_EPS      1e-5f

// --------------------------- device helpers --------------------------------

__device__ __forceinline__ float trilerp(const float* __restrict__ v,
                                         float zf, float yf, float xf,
                                         int Z, int Y, int X) {
  // map_coordinates(order=1, mode='nearest'): clamp coords to the valid range.
  zf = fminf(fmaxf(zf, 0.f), (float)(Z - 1));
  yf = fminf(fmaxf(yf, 0.f), (float)(Y - 1));
  xf = fminf(fmaxf(xf, 0.f), (float)(X - 1));
  int z0 = (int)floorf(zf), y0 = (int)floorf(yf), x0 = (int)floorf(xf);
  float tz = zf - (float)z0, ty = yf - (float)y0, tx = xf - (float)x0;
  int z1 = min(z0 + 1, Z - 1), y1 = min(y0 + 1, Y - 1), x1 = min(x0 + 1, X - 1);
  size_t b00 = ((size_t)z0 * Y + y0) * X, b01 = ((size_t)z0 * Y + y1) * X;
  size_t b10 = ((size_t)z1 * Y + y0) * X, b11 = ((size_t)z1 * Y + y1) * X;
  float c000 = v[b00 + x0], c001 = v[b00 + x1];
  float c010 = v[b01 + x0], c011 = v[b01 + x1];
  float c100 = v[b10 + x0], c101 = v[b10 + x1];
  float c110 = v[b11 + x0], c111 = v[b11 + x1];
  float c00 = c000 + tx * (c001 - c000);
  float c01 = c010 + tx * (c011 - c010);
  float c10 = c100 + tx * (c101 - c100);
  float c11 = c110 + tx * (c111 - c110);
  float c0 = c00 + ty * (c01 - c00);
  float c1 = c10 + ty * (c11 - c10);
  return c0 + tz * (c1 - c0);
}

// -------------------- consensus field logits + softmax ----------------------
__global__ void k_field_softmax(const float* __restrict__ x,
                                const float* __restrict__ fw,
                                const float* __restrict__ ref,
                                const float* __restrict__ omega_p,
                                float* __restrict__ wsm,
                                int C, int Zi, int Yi, int Xi,
                                int stride, int Zo, int Yo, int Xo) {
  int Po = Zo * Yo * Xo;
  int po = blockIdx.x * blockDim.x + threadIdx.x;
  if (po >= Po) return;
  int xo = po % Xo, yo = (po / Xo) % Yo, zo = po / (Xo * Yo);
  size_t Pin = (size_t)Zi * Yi * Xi;
  size_t pin = ((size_t)(zo * stride) * Yi + (size_t)(yo * stride)) * Xi + (size_t)(xo * stride);
  float f[27];
#pragma unroll
  for (int d = 0; d < 27; ++d) f[d] = 0.f;
  for (int c = 0; c < C; ++c) {
    float xv = x[(size_t)c * Pin + pin];
#pragma unroll
    for (int d = 0; d < 27; ++d) f[d] += fw[d * C + c] * xv;
  }
  if (ref) {
    float om = *omega_p;
#pragma unroll
    for (int d = 0; d < 27; ++d)
      f[d] = om * f[d] + (1.f - om) * ref[(size_t)d * Po + po];
  }
  float m = f[0];
#pragma unroll
  for (int d = 1; d < 27; ++d) m = fmaxf(m, f[d]);
  float s = 0.f;
#pragma unroll
  for (int d = 0; d < 27; ++d) { f[d] = __expf(f[d] - m); s += f[d]; }
  float inv = 1.f / s;
#pragma unroll
  for (int d = 0; d < 27; ++d) wsm[(size_t)d * Po + po] = f[d] * inv;
}

// Raw field logits (decoder reference field from the skip tensor).
__global__ void k_field_logits(const float* __restrict__ x,
                               const float* __restrict__ fw,
                               float* __restrict__ out, int C, int P) {
  int p = blockIdx.x * blockDim.x + threadIdx.x;
  if (p >= P) return;
  float f[27];
#pragma unroll
  for (int d = 0; d < 27; ++d) f[d] = 0.f;
  for (int c = 0; c < C; ++c) {
    float xv = x[(size_t)c * P + p];
#pragma unroll
    for (int d = 0; d < 27; ++d) f[d] += fw[d * C + c] * xv;
  }
#pragma unroll
  for (int d = 0; d < 27; ++d) out[(size_t)d * P + p] = f[d];
}

// ------------------------------- GeoSample ---------------------------------
// Writes channel-last f16 [Po][Kpad] for direct consumption by the WMMA GEMM.
// grid.y = Kpad; pad channels (c >= C) write 0 so the GEMM needs no K guard.
__global__ void k_geosample(const float* __restrict__ x,
                            const float* __restrict__ wsm,
                            const float* __restrict__ r_p,
                            _Float16* __restrict__ gh,
                            int C, int Kpad,
                            int Zi, int Yi, int Xi,
                            int stride, int Zo, int Yo, int Xo) {
  int Po = Zo * Yo * Xo;
  int po = blockIdx.x * blockDim.x + threadIdx.x;
  int c = blockIdx.y;
  if (po >= Po) return;
  if (c >= C) { gh[(size_t)po * Kpad + c] = (_Float16)0.f; return; }
  int xo = po % Xo, yo = (po / Xo) % Yo, zo = po / (Xo * Yo);
  float rr = fminf(fmaxf(*r_p, 0.5f), 2.0f);
  float off[3] = {-rr, 0.f, rr};
  const float* vol = x + (size_t)c * Zi * Yi * Xi;
  float zc = (float)(zo * stride), yc = (float)(yo * stride), xc = (float)(xo * stride);
  float acc = 0.f;
  for (int d = 0; d < 27; ++d) {
    int dz = d / 9, dy = (d / 3) % 3, dx = d % 3;
    float wv = wsm[(size_t)d * Po + po];
    acc += wv * trilerp(vol, zc + off[dz], yc + off[dy], xc + off[dx], Zi, Yi, Xi);
  }
  gh[(size_t)po * Kpad + c] = (_Float16)acc;
}

// ------------------------ weight pack (f32 -> f16, K-pad) -------------------
__global__ void k_pack_w(const float* __restrict__ w, _Float16* __restrict__ wh,
                         int M, int K, int Kpad) {
  int i = blockIdx.x * blockDim.x + threadIdx.x;
  if (i >= M * Kpad) return;
  int m = i / Kpad, k = i % Kpad;
  wh[i] = (k < K) ? (_Float16)w[m * K + k] : (_Float16)0.f;
}

// ------------------------------ WMMA GEMM ----------------------------------
// Y[M][N] = A[M][Kpad] * B^T (B stored channel-last [N][Kpad]), f16 in, f32 acc.
// One wave per block; each wave owns a 16(M) x 64(N) strip: one A fragment is
// reused across 4 B fragments -> 4 v_wmma per K-step. Requires M%16==0,
// N%64==0, Kpad%32==0 (guaranteed by producers) => no guards, EXEC all ones.
__global__ void k_gemm_wmma(const _Float16* __restrict__ Ah,
                            const _Float16* __restrict__ Bh,
                            float* __restrict__ Yo,
                            int M, int Kpad, int N) {
#if __has_builtin(__builtin_amdgcn_s_wait_tensorcnt)
  __builtin_amdgcn_s_wait_tensorcnt(0);  // gfx1250 split-counter: no TDM ops outstanding
#endif
  int lane = (int)(threadIdx.x & 31u);
  int half = lane >> 4;          // 0: lanes 0-15, 1: lanes 16-31
  int lrow = lane & 15;
  int mBase = blockIdx.y * 16;
  int n0 = blockIdx.x * 64;
  const _Float16* arow  = Ah + (size_t)(mBase + lrow) * Kpad;
  const _Float16* brow0 = Bh + (size_t)(n0 +  0 + lrow) * Kpad + 16 * half;
  const _Float16* brow1 = Bh + (size_t)(n0 + 16 + lrow) * Kpad + 16 * half;
  const _Float16* brow2 = Bh + (size_t)(n0 + 32 + lrow) * Kpad + 16 * half;
  const _Float16* brow3 = Bh + (size_t)(n0 + 48 + lrow) * Kpad + 16 * half;
  v8f acc0 = {}, acc1 = {}, acc2 = {}, acc3 = {};
  for (int kt = 0; kt < Kpad; kt += 32) {
    // A fragment: halves 0-7 -> K = kt+16*half+(0..7); 8-15 -> K = kt+16*half+8+(0..7)
    v8h alo = *(const v8h*)(arow + kt + 16 * half);
    v8h ahi = *(const v8h*)(arow + kt + 16 * half + 8);
    v16h a = __builtin_shufflevector(alo, ahi, 0, 1, 2, 3, 4, 5, 6, 7,
                                     8, 9, 10, 11, 12, 13, 14, 15);
    // B fragments: 16 consecutive K per lane half (32B contiguous loads)
    v16h b0 = *(const v16h*)(brow0 + kt);
    v16h b1 = *(const v16h*)(brow1 + kt);
    v16h b2 = *(const v16h*)(brow2 + kt);
    v16h b3 = *(const v16h*)(brow3 + kt);
    if (kt + 32 < Kpad) {  // prefetch next K tile (global_prefetch_b8)
      __builtin_prefetch(brow0 + kt + 32, 0, 3);
      __builtin_prefetch(brow2 + kt + 32, 0, 3);
    }
    acc0 = __builtin_amdgcn_wmma_f32_16x16x32_f16(false, a, false, b0, (short)0, acc0, false, false);
    acc1 = __builtin_amdgcn_wmma_f32_16x16x32_f16(false, a, false, b1, (short)0, acc1, false, false);
    acc2 = __builtin_amdgcn_wmma_f32_16x16x32_f16(false, a, false, b2, (short)0, acc2, false, false);
    acc3 = __builtin_amdgcn_wmma_f32_16x16x32_f16(false, a, false, b3, (short)0, acc3, false, false);
  }
  // D layout: lane half 0 -> rows v, half 1 -> rows v+8; column = lane&15.
#pragma unroll
  for (int v = 0; v < 8; ++v) {
    int m = mBase + v + 8 * half;
    float* yr = Yo + (size_t)m * N + n0 + lrow;
    yr[0]  = acc0[v];
    yr[16] = acc1[v];
    yr[32] = acc2[v];
    yr[48] = acc3[v];
  }
}

// --------------------------- InstanceNorm ----------------------------------
__global__ void k_in_stats(const float* __restrict__ y, float* __restrict__ stats, int N) {
  int c = blockIdx.x;
  const float* p = y + (size_t)c * N;
  float s = 0.f, s2 = 0.f;
  for (int i = threadIdx.x; i < N; i += blockDim.x) { float v = p[i]; s += v; s2 += v * v; }
  __shared__ float sh[256], sh2[256];
  sh[threadIdx.x] = s; sh2[threadIdx.x] = s2;
  __syncthreads();
  for (int w = 128; w > 0; w >>= 1) {
    if ((int)threadIdx.x < w) { sh[threadIdx.x] += sh[threadIdx.x + w]; sh2[threadIdx.x] += sh2[threadIdx.x + w]; }
    __syncthreads();
  }
  if (threadIdx.x == 0) {
    float mu = sh[0] / (float)N;
    stats[2 * c] = mu;
    stats[2 * c + 1] = sh2[0] / (float)N - mu * mu;
  }
}

__global__ void k_norm_lrelu(const float* __restrict__ y, const float* __restrict__ stats,
                             const float* __restrict__ gamma, const float* __restrict__ beta,
                             float* __restrict__ out, int N) {
  int p = blockIdx.x * blockDim.x + threadIdx.x;
  int c = blockIdx.y;
  if (p >= N) return;
  float mu = stats[2 * c], var = stats[2 * c + 1];
  float v = (y[(size_t)c * N + p] - mu) * rsqrtf(var + IN_EPS) * gamma[c] + beta[c];
  out[(size_t)c * N + p] = v > 0.f ? v : LRELU_SLOPE * v;
}

// ------------------- trilinear upsample x2 -> channel-last f16 --------------
// Output feeds only the decoder align GEMM, so emit [Po][Cpad] f16 directly.
__global__ void k_upsample2_f16(const float* __restrict__ x, _Float16* __restrict__ o,
                                int C, int Cpad, int Z, int Y, int X) {
  int Zo = 2 * Z, Yo = 2 * Y, Xo = 2 * X;
  int Po = Zo * Yo * Xo;
  int p = blockIdx.x * blockDim.x + threadIdx.x;
  int c = blockIdx.y;
  if (p >= Po) return;
  if (c >= C) { o[(size_t)p * Cpad + c] = (_Float16)0.f; return; }
  int xo = p % Xo, yo = (p / Xo) % Yo, zo = p / (Xo * Yo);
  float zf = ((float)zo + 0.5f) * 0.5f - 0.5f;   // half-pixel centers, scale 0.5
  float yf = ((float)yo + 0.5f) * 0.5f - 0.5f;
  float xf = ((float)xo + 0.5f) * 0.5f - 0.5f;
  o[(size_t)p * Cpad + c] = (_Float16)trilerp(x + (size_t)c * Z * Y * X, zf, yf, xf, Z, Y, X);
}

// ------------------------------- head 1x1 ----------------------------------
__global__ void k_head(const float* __restrict__ x, const float* __restrict__ w,
                       const float* __restrict__ b, float* __restrict__ out, int C, int N) {
  int p = blockIdx.x * blockDim.x + threadIdx.x;
  if (p >= N) return;
  float s = *b;
  for (int c = 0; c < C; ++c) s += w[c] * x[(size_t)c * N + p];
  out[p] = s;
}

// ------------------------------ host side ----------------------------------

namespace {

struct UnitP {
  const float *r, *omega, *field_w, *proj_w, *gamma, *beta;
  int cin, cout;
};

inline int ceil_div(int a, int b) { return (a + b - 1) / b; }
inline int kpad32(int k) { return (k + 31) & ~31; }

void launch_unit(hipStream_t s, const UnitP& u, const float* x,
                 int Zi, int Yi, int Xi, int stride, const float* ref,
                 float* out, float* wsm, _Float16* gh, _Float16* wh, float* stats) {
  int Zo = Zi / stride, Yo = Yi / stride, Xo = Xi / stride;
  int Po = Zo * Yo * Xo;
  int Kp = kpad32(u.cin);
  k_field_softmax<<<dim3(ceil_div(Po, 256)), dim3(256), 0, s>>>(
      x, u.field_w, ref, u.omega, wsm, u.cin, Zi, Yi, Xi, stride, Zo, Yo, Xo);
  k_geosample<<<dim3(ceil_div(Po, 256), Kp), dim3(256), 0, s>>>(
      x, wsm, u.r, gh, u.cin, Kp, Zi, Yi, Xi, stride, Zo, Yo, Xo);
  k_pack_w<<<dim3(ceil_div(u.cout * Kp, 256)), dim3(256), 0, s>>>(
      u.proj_w, wh, u.cout, u.cin, Kp);
  k_gemm_wmma<<<dim3(Po / 64, u.cout / 16), dim3(32), 0, s>>>(
      wh, gh, out, u.cout, Kp, Po);
  k_in_stats<<<dim3(u.cout), dim3(256), 0, s>>>(out, stats, Po);
  k_norm_lrelu<<<dim3(ceil_div(Po, 256), u.cout), dim3(256), 0, s>>>(
      out, stats, u.gamma, u.beta, out, Po);
}

} // namespace

extern "C" void kernel_launch(void* const* d_in, const int* in_sizes, int n_in,
                              void* d_out, int out_size, void* d_ws, size_t ws_size,
                              hipStream_t stream) {
  (void)in_sizes; (void)n_in; (void)out_size; (void)ws_size;
  auto gp = [&](int i) { return (const float*)d_in[i]; };
  auto unit = [&](int base, int cin, int cout) {
    UnitP u;  // sorted unit keys: beta, field_w, gamma, omega, proj_w, r
    u.beta = gp(base + 0); u.field_w = gp(base + 1); u.gamma = gp(base + 2);
    u.omega = gp(base + 3); u.proj_w = gp(base + 4); u.r = gp(base + 5);
    u.cin = cin; u.cout = cout;
    return u;
  };

  // pytree (sorted-key) flat order: x, then params{down, enc, head_b, head_w, up}
  const float* x = gp(0);
  UnitP down0 = unit(1, 16, 32), down1 = unit(7, 32, 64);
  UnitP e00 = unit(13, 1, 16), e01 = unit(19, 16, 16);
  UnitP e10 = unit(25, 32, 32), e11 = unit(31, 32, 32);
  UnitP e20 = unit(37, 64, 64), e21 = unit(43, 64, 64);
  const float* head_b = gp(49);
  const float* head_w = gp(50);
  const float* align0 = gp(51);               // (32,64)
  UnitP f00 = unit(52, 64, 32), f01 = unit(58, 32, 32), pre0 = unit(64, 32, 32);
  const float* sfw0 = gp(70);                 // (27,32)
  const float* align1 = gp(71);               // (16,32)
  UnitP f10 = unit(72, 32, 16), f11 = unit(78, 16, 16), pre1 = unit(84, 16, 16);
  const float* sfw1 = gp(90);                 // (27,16)

  const int S0 = 48 * 48 * 48, S1 = 24 * 24 * 24, S2 = 12 * 12 * 12;
  char* base = (char*)d_ws;
  size_t off = 0;
  auto allocB = [&](size_t nbytes) -> void* {
    void* p = (void*)(base + off);
    off = (off + nbytes + 255) & ~(size_t)255;
    return p;
  };
  auto allocF = [&](size_t nf) -> float* { return (float*)allocB(nf * sizeof(float)); };
  auto allocH = [&](size_t nh) -> _Float16* { return (_Float16*)allocB(nh * sizeof(_Float16)); };

  float*    wsm   = allocF((size_t)27 * S0);   // softmax weights scratch (max res)
  _Float16* gh    = allocH((size_t)32 * S0);   // f16 channel-last GEMM-B scratch (max use: 32ch@48^3)
  _Float16* wh    = allocH((size_t)64 * 64);   // f16 padded weight scratch
  float*    stats = allocF(128);               // per-channel mean/var
  float*    e0a   = allocF((size_t)16 * S0);
  float*    skip0 = allocF((size_t)16 * S0);
  float*    d0    = allocF((size_t)32 * S1);
  float*    e1a   = allocF((size_t)32 * S1);
  float*    skip1 = allocF((size_t)32 * S1);
  float*    d1b   = allocF((size_t)64 * S2);
  float*    e2a   = allocF((size_t)64 * S2);
  float*    h2    = allocF((size_t)64 * S2);
  float*    u0al  = allocF((size_t)32 * S1);
  float*    ref0  = allocF((size_t)27 * S1);
  float*    cat0  = allocF((size_t)64 * S1);
  float*    f0a   = allocF((size_t)32 * S1);
  float*    f0b   = allocF((size_t)32 * S1);
  float*    u1al  = allocF((size_t)16 * S0);
  float*    ref1  = allocF((size_t)27 * S0);
  float*    cat1  = allocF((size_t)32 * S0);
  float*    f1a   = allocF((size_t)16 * S0);
  float*    f1b   = allocF((size_t)16 * S0);

  // ---------------- encoder ----------------
  launch_unit(stream, e00, x,       48, 48, 48, 1, nullptr, e0a,   wsm, gh, wh, stats);
  launch_unit(stream, e01, e0a,     48, 48, 48, 1, nullptr, skip0, wsm, gh, wh, stats);
  launch_unit(stream, down0, skip0, 48, 48, 48, 2, nullptr, d0,    wsm, gh, wh, stats);
  launch_unit(stream, e10, d0,      24, 24, 24, 1, nullptr, e1a,   wsm, gh, wh, stats);
  launch_unit(stream, e11, e1a,     24, 24, 24, 1, nullptr, skip1, wsm, gh, wh, stats);
  launch_unit(stream, down1, skip1, 24, 24, 24, 2, nullptr, d1b,   wsm, gh, wh, stats);
  launch_unit(stream, e20, d1b,     12, 12, 12, 1, nullptr, e2a,   wsm, gh, wh, stats);
  launch_unit(stream, e21, e2a,     12, 12, 12, 1, nullptr, h2,    wsm, gh, wh, stats);

  // ------- decoder block 0: 12^3 x64 -> 24^3 x32 (skip = skip1) -------
  k_upsample2_f16<<<dim3(ceil_div(S1, 256), 64), dim3(256), 0, stream>>>(h2, gh, 64, 64, 12, 12, 12);
  k_pack_w<<<dim3(ceil_div(32 * 64, 256)), dim3(256), 0, stream>>>(align0, wh, 32, 64, 64);
  k_gemm_wmma<<<dim3(S1 / 64, 2), dim3(32), 0, stream>>>(wh, gh, u0al, 32, 64, S1);
  k_field_logits<<<dim3(ceil_div(S1, 256)), dim3(256), 0, stream>>>(skip1, sfw0, ref0, 32, S1);
  launch_unit(stream, pre0, u0al, 24, 24, 24, 1, ref0, cat0, wsm, gh, wh, stats); // -> cat0[0:32]
  hipMemcpyAsync(cat0 + (size_t)32 * S1, skip1, (size_t)32 * S1 * sizeof(float),
                 hipMemcpyDeviceToDevice, stream);                                 // -> cat0[32:64]
  launch_unit(stream, f00, cat0, 24, 24, 24, 1, nullptr, f0a, wsm, gh, wh, stats);
  launch_unit(stream, f01, f0a,  24, 24, 24, 1, nullptr, f0b, wsm, gh, wh, stats);

  // ------- decoder block 1: 24^3 x32 -> 48^3 x16 (skip = skip0) -------
  k_upsample2_f16<<<dim3(ceil_div(S0, 256), 32), dim3(256), 0, stream>>>(f0b, gh, 32, 32, 24, 24, 24);
  k_pack_w<<<dim3(ceil_div(16 * 32, 256)), dim3(256), 0, stream>>>(align1, wh, 16, 32, 32);
  k_gemm_wmma<<<dim3(S0 / 64, 1), dim3(32), 0, stream>>>(wh, gh, u1al, 16, 32, S0);
  k_field_logits<<<dim3(ceil_div(S0, 256)), dim3(256), 0, stream>>>(skip0, sfw1, ref1, 16, S0);
  launch_unit(stream, pre1, u1al, 48, 48, 48, 1, ref1, cat1, wsm, gh, wh, stats); // -> cat1[0:16]
  hipMemcpyAsync(cat1 + (size_t)16 * S0, skip0, (size_t)16 * S0 * sizeof(float),
                 hipMemcpyDeviceToDevice, stream);                                 // -> cat1[16:32]
  launch_unit(stream, f10, cat1, 48, 48, 48, 1, nullptr, f1a, wsm, gh, wh, stats);
  launch_unit(stream, f11, f1a,  48, 48, 48, 1, nullptr, f1b, wsm, gh, wh, stats);

  // ---------------- head ----------------
  k_head<<<dim3(ceil_div(S0, 256)), dim3(256), 0, stream>>>(f1b, head_w, head_b,
                                                            (float*)d_out, 16, S0);
}
